// GsGLstm_24532853195501
// MI455X (gfx1250) — compile-verified, matrix operand order checked by reading, not run on hardware
//
#include <hip/hip_runtime.h>
#include <hip/hip_bf16.h>
#include <stdint.h>
#include <stddef.h>

// Problem constants (match reference)
#define B_   8
#define N_   1024
#define K_   16
#define D_   256
#define G_   4          // gates i,o,f,cell
#define KT_  8          // D / 32  (K-tiles per WMMA chain)
#define ET_  16         // D / 16  (16-wide output column tiles)
#define MROWS_ 64       // rows per block (4 WMMA M-tiles)

typedef __attribute__((ext_vector_type(16))) __bf16 v16bf;
typedef __attribute__((ext_vector_type(8)))  float  v8f;
typedef __attribute__((ext_vector_type(4)))  float  f32x4;

union Frag { v16bf v; f32x4 q[2]; };

__device__ __forceinline__ unsigned short f2bf(float x) {
    unsigned int u = __float_as_uint(x);
    u += 0x7FFFu + ((u >> 16) & 1u);          // round-to-nearest-even
    return (unsigned short)(u >> 16);
}
__device__ __forceinline__ float sigmoidf_(float x) {
    return 1.0f / (1.0f + __expf(-x));
}

// ---------------------------------------------------------------------------
// Repack f32 weight [G][D][D] (pre[g,b,n,e] = sum_d A[d] * W[g][d][e]) into the
// CDNA5 16-bit B-matrix (32x16) fragment layout, bf16:
//   packed[((g*KT + kt)*ET + et)*32 + lane][j],  lane: col n = lane&15,
//   kbase = (lane>>4)*8, j<8 -> K = kt*32+kbase+j, j>=8 -> K = kt*32+kbase+8+j
// ---------------------------------------------------------------------------
__global__ void pack_weight_kernel(const float* __restrict__ W,
                                   unsigned short* __restrict__ P) {
    int t = blockIdx.x * blockDim.x + threadIdx.x;   // over G*KT*ET*32*16
    int j    = t & 15;
    int lane = (t >> 4) & 31;
    int et   = (t >> 9) & 15;
    int kt   = (t >> 13) & 7;
    int g    = t >> 16;
    int kb   = (lane >> 4) << 3;
    int k    = kt * 32 + kb + (j < 8 ? j : 8 + j);
    int col  = et * 16 + (lane & 15);
    P[t] = f2bf(W[((size_t)g * D_ + k) * D_ + col]);
}

__global__ void f32_to_bf16_kernel(const float* __restrict__ src,
                                   unsigned short* __restrict__ dst, int n) {
    int t = blockIdx.x * blockDim.x + threadIdx.x;
    if (t < n) dst[t] = f2bf(src[t]);
}

// ---------------------------------------------------------------------------
// Masked neighbor aggregation: one block per (b,n), one thread per channel e.
// h is L2-resident (8.4 MB) so the 16-way gather hits the 192 MB L2.
// Emits bf16 rows ready for WMMA A-tile staging.
// ---------------------------------------------------------------------------
__global__ void gather_kernel(const float* __restrict__ h,
                              const int*   __restrict__ in_nodes,
                              const float* __restrict__ in_mask,
                              const int*   __restrict__ out_nodes,
                              const float* __restrict__ out_mask,
                              const float* __restrict__ node_mask,
                              unsigned short* __restrict__ hin_bf,
                              unsigned short* __restrict__ hout_bf) {
    int bn = blockIdx.x;              // 0 .. B*N-1
    int e  = threadIdx.x;             // 0 .. 255
    int b  = bn >> 10;                // / N_
    const float* hb = h + (size_t)b * N_ * D_;
    float ai = 0.f, ao = 0.f;
    #pragma unroll
    for (int k = 0; k < K_; ++k) {
        int   ii = in_nodes [bn * K_ + k];
        float mi = in_mask  [bn * K_ + k];
        ai += mi * hb[(size_t)ii * D_ + e];
        int   oi = out_nodes[bn * K_ + k];
        float mo = out_mask [bn * K_ + k];
        ao += mo * hb[(size_t)oi * D_ + e];
    }
    float nm = node_mask[bn];
    hin_bf [(size_t)bn * D_ + e] = f2bf(ai * nm);
    hout_bf[(size_t)bn * D_ + e] = f2bf(ao * nm);
}

// ---------------------------------------------------------------------------
// Dual bf16 GEMM: acc[g] = A1 @ Bp1[g] + A2 @ Bp2[g]   (A: [B*N][256] bf16)
// Block = 64 rows x full 1024 output cols (4 gates * 256), 16 waves.
// Wave w owns the 16-wide e-tile  e = [w*16, w*16+16)  for ALL 4 gates and
// all 4 M-tiles -> each B-fragment pair is reused by 4 A-fragments (8 WMMAs
// per 2 global fragment loads), cutting L2 weight traffic 4x vs 16-row
// blocks and keeping the LSTM gate math wave-local.
// MODE 0: epilogue adds bias, stores xpre.
// MODE 1: epilogue adds xpre, applies LSTM gates, updates c and h in place.
// ---------------------------------------------------------------------------
template<int MODE>
__global__ void __launch_bounds__(512)
dual_gemm_kernel(const unsigned short* __restrict__ A1,
                 const unsigned short* __restrict__ A2,
                 const unsigned short* __restrict__ Bp1,
                 const unsigned short* __restrict__ Bp2,
                 const float* __restrict__ bias,       // MODE 0: [G][D]
                 const float* __restrict__ xpre,       // MODE 1: [B*N][G*D]
                 float* __restrict__ xpre_out,         // MODE 0
                 float* __restrict__ c_buf,            // MODE 1 (in/out)
                 float* __restrict__ h_buf,            // MODE 1 (out)
                 const float* __restrict__ node_mask)  // MODE 1
{
    __shared__ unsigned short ldsA1[MROWS_ * D_];   // 32 KB
    __shared__ unsigned short ldsA2[MROWS_ * D_];   // 32 KB

    int tid  = threadIdx.x;
    int row0 = blockIdx.x * MROWS_;

    // Stage both 64x256 bf16 A tiles into LDS (coalesced b128 loads)
    {
        const f32x4* s1 = (const f32x4*)(A1 + (size_t)row0 * D_);
        const f32x4* s2 = (const f32x4*)(A2 + (size_t)row0 * D_);
        f32x4* d1 = (f32x4*)ldsA1;
        f32x4* d2 = (f32x4*)ldsA2;
        #pragma unroll
        for (int i = 0; i < 4; ++i) {             // 2048 f32x4 per tile / 512 thr
            d1[tid + i * 512] = s1[tid + i * 512];
            d2[tid + i * 512] = s2[tid + i * 512];
        }
    }
    __syncthreads();

    int wave = tid >> 5;           // 0..15 : this wave's e-tile
    int lane = tid & 31;
    int n    = lane & 15;          // output column within tile / A row
    int hi   = lane >> 4;

    v8f zero = {};
    v8f acc[G_][4];                // [gate][mtile] : 128 VGPRs
    #pragma unroll
    for (int g = 0; g < G_; ++g)
        #pragma unroll
        for (int mt = 0; mt < 4; ++mt) acc[g][mt] = zero;

    #pragma unroll
    for (int kt = 0; kt < KT_; ++kt) {
        // A fragments (16-bit A 16x32 layout): row m = lane&15,
        // K chunks at kbase and kbase+16, kbase = kt*32 + (lane>>4)*8
        Frag fa1[4], fa2[4];
        int kb = kt * 32 + (hi << 3);
        #pragma unroll
        for (int mt = 0; mt < 4; ++mt) {
            const f32x4* p1 = (const f32x4*)(&ldsA1[(mt * 16 + n) * D_ + kb]);
            const f32x4* p2 = (const f32x4*)(&ldsA2[(mt * 16 + n) * D_ + kb]);
            fa1[mt].q[0] = p1[0]; fa1[mt].q[1] = p1[2];   // +32B = +16 bf16 cols
            fa2[mt].q[0] = p2[0]; fa2[mt].q[1] = p2[2];
        }
        #pragma unroll
        for (int g = 0; g < G_; ++g) {
            Frag fb1, fb2;
            size_t base =
                ((((size_t)(g * KT_ + kt)) * ET_ + wave) * 32 + lane) * 16;
            const f32x4* q1 = (const f32x4*)(Bp1 + base);
            const f32x4* q2 = (const f32x4*)(Bp2 + base);
            fb1.q[0] = q1[0]; fb1.q[1] = q1[1];
            fb2.q[0] = q2[0]; fb2.q[1] = q2[1];
            #pragma unroll
            for (int mt = 0; mt < 4; ++mt) {
                acc[g][mt] = __builtin_amdgcn_wmma_f32_16x16x32_bf16(
                    false, fa1[mt].v, false, fb1.v, (short)0, acc[g][mt], false, false);
                acc[g][mt] = __builtin_amdgcn_wmma_f32_16x16x32_bf16(
                    false, fa2[mt].v, false, fb2.v, (short)0, acc[g][mt], false, false);
            }
        }
    }

    // C/D layout: VGPR r -> row m = r + (lane>>4)*8, col = wave*16 + (lane&15)
    int e = wave * 16 + n;
    if (MODE == 0) {
        #pragma unroll
        for (int g = 0; g < G_; ++g) {
            float bv = bias[g * D_ + e];
            #pragma unroll
            for (int mt = 0; mt < 4; ++mt)
                #pragma unroll
                for (int r = 0; r < 8; ++r) {
                    size_t gr = (size_t)(row0 + mt * 16 + r + hi * 8);
                    xpre_out[gr * (G_ * D_) + g * D_ + e] = acc[g][mt][r] + bv;
                }
        }
    } else {
        #pragma unroll
        for (int mt = 0; mt < 4; ++mt)
            #pragma unroll
            for (int r = 0; r < 8; ++r) {
                size_t gr = (size_t)(row0 + mt * 16 + r + hi * 8);
                const float* xp = xpre + gr * (G_ * D_);
                float pi = acc[0][mt][r] + xp[0 * D_ + e];
                float po = acc[1][mt][r] + xp[1 * D_ + e];
                float pf = acc[2][mt][r] + xp[2 * D_ + e];
                float pg = acc[3][mt][r] + xp[3 * D_ + e];
                float ig = sigmoidf_(pi);
                float og = sigmoidf_(po);
                float fg = sigmoidf_(pf);
                float gg = tanhf(pg);
                float nm = node_mask[gr];
                float cn = (fg * c_buf[gr * D_ + e] + ig * gg) * nm;
                float hn = og * tanhf(cn) * nm;
                c_buf[gr * D_ + e] = cn;
                h_buf[gr * D_ + e] = hn;
            }
    }
}

// ---------------------------------------------------------------------------
extern "C" void kernel_launch(void* const* d_in, const int* in_sizes, int n_in,
                              void* d_out, int out_size, void* d_ws, size_t ws_size,
                              hipStream_t stream) {
    (void)in_sizes; (void)n_in; (void)out_size; (void)ws_size;

    const float* h0        = (const float*)d_in[0];
    const float* c0        = (const float*)d_in[1];
    const float* x_in      = (const float*)d_in[2];
    const float* x_out     = (const float*)d_in[3];
    const float* W_in      = (const float*)d_in[4];
    const float* U_in      = (const float*)d_in[5];
    const float* W_out     = (const float*)d_in[6];
    const float* U_out     = (const float*)d_in[7];
    const float* bvec      = (const float*)d_in[8];
    const float* in_mask   = (const float*)d_in[9];
    const float* out_mask  = (const float*)d_in[10];
    const float* node_mask = (const float*)d_in[11];
    const int*   in_nodes  = (const int*)d_in[12];
    const int*   out_nodes = (const int*)d_in[13];
    // d_in[14] = num_layers; fixed at 3 in the reference.

    const size_t hbytes   = (size_t)B_ * N_ * D_ * sizeof(float);          // 8 MiB
    const size_t hbf      = (size_t)B_ * N_ * D_ * sizeof(unsigned short); // 4 MiB
    const size_t xpbytes  = (size_t)B_ * N_ * G_ * D_ * sizeof(float);     // 32 MiB
    const size_t wpkbytes = (size_t)G_ * KT_ * ET_ * 32 * 16 * sizeof(unsigned short); // 512 KiB

    char* ws = (char*)d_ws;
    float*          h_ws  = (float*)ws;            ws += hbytes;
    float*          c_ws  = (float*)ws;            ws += hbytes;
    float*          xpre  = (float*)ws;            ws += xpbytes;
    unsigned short* hin   = (unsigned short*)ws;   ws += hbf;
    unsigned short* hout  = (unsigned short*)ws;   ws += hbf;
    unsigned short* xinb  = (unsigned short*)ws;   ws += hbf;
    unsigned short* xoutb = (unsigned short*)ws;   ws += hbf;
    unsigned short* Wp_in = (unsigned short*)ws;   ws += wpkbytes;
    unsigned short* Up_in = (unsigned short*)ws;   ws += wpkbytes;
    unsigned short* Wp_out= (unsigned short*)ws;   ws += wpkbytes;
    unsigned short* Up_out= (unsigned short*)ws;   ws += wpkbytes;

    // State copies (inputs must stay untouched)
    hipMemcpyAsync(h_ws, h0, hbytes, hipMemcpyDeviceToDevice, stream);
    hipMemcpyAsync(c_ws, c0, hbytes, hipMemcpyDeviceToDevice, stream);

    // Weight repack -> bf16 WMMA B-fragment layout
    const int packN = G_ * KT_ * ET_ * 32 * 16;   // 262144
    pack_weight_kernel<<<packN / 256, 256, 0, stream>>>(W_in,  Wp_in);
    pack_weight_kernel<<<packN / 256, 256, 0, stream>>>(U_in,  Up_in);
    pack_weight_kernel<<<packN / 256, 256, 0, stream>>>(W_out, Wp_out);
    pack_weight_kernel<<<packN / 256, 256, 0, stream>>>(U_out, Up_out);

    // x tensors -> bf16
    const int xn = B_ * N_ * D_;
    f32_to_bf16_kernel<<<xn / 256, 256, 0, stream>>>(x_in,  xinb,  xn);
    f32_to_bf16_kernel<<<xn / 256, 256, 0, stream>>>(x_out, xoutb, xn);

    const int gemmBlocks = (B_ * N_) / MROWS_;   // 128

    // Layer-invariant xpre = x_in@W_in + x_out@W_out + b (computed once)
    dual_gemm_kernel<0><<<gemmBlocks, 512, 0, stream>>>(
        xinb, xoutb, Wp_in, Wp_out, bvec,
        nullptr, xpre, nullptr, nullptr, nullptr);

    for (int layer = 0; layer < 3; ++layer) {
        gather_kernel<<<B_ * N_, 256, 0, stream>>>(
            h_ws, in_nodes, in_mask, out_nodes, out_mask, node_mask, hin, hout);
        dual_gemm_kernel<1><<<gemmBlocks, 512, 0, stream>>>(
            hin, hout, Up_in, Up_out, nullptr,
            xpre, nullptr, c_ws, h_ws, node_mask);
    }

    hipMemcpyAsync(d_out, h_ws, hbytes, hipMemcpyDeviceToDevice, stream);
}
